// ClusterFeatureBranch_13889924235275
// MI455X (gfx1250) — compile-verified, matrix operand order checked by reading, not run on hardware
//
#include <hip/hip_runtime.h>
#include <math.h>

// ---------- types for WMMA / TDM ----------
typedef __attribute__((ext_vector_type(16))) _Float16     v16h;
typedef __attribute__((ext_vector_type(8)))  float        v8f;
typedef __attribute__((ext_vector_type(8)))  unsigned int v8u;
typedef __attribute__((ext_vector_type(4)))  unsigned int v4u;
typedef __attribute__((ext_vector_type(8)))  int          v8i;
typedef __attribute__((ext_vector_type(4)))  int          v4i;

#define HF   512   // hidden width (4 heads x 128)
#define NH   4     // heads
#define FH   128   // per-head channels
#define COUT 896   // 512 topo + 256 pca + 128 scale

static inline int cdiv_ll(long long a, long long b){ return (int)((a + b - 1) / b); }

// ---------------------------------------------------------------------------
// utility
// ---------------------------------------------------------------------------
__global__ void k_fill_zero(float* __restrict__ p, long long n) {
  long long i = (long long)blockIdx.x * blockDim.x + threadIdx.x;
  if (i < n) p[i] = 0.f;
}

// Pack a KxN f32 weight matrix into the wave32 WMMA B-fragment layout
// (16-bit B matrix, 16x16x32): Bp[((nt*KT + kt)*32 + lane)*8 + v] holds the
// dword {B[k,n], B[k+1,n]} with n = nt*16 + lane%16,
// k = kt*32 + (v/4)*16 + (lane/16)*8 + (v%4)*2.
__global__ void k_pack_b(const float* __restrict__ B, unsigned int* __restrict__ Bp,
                         int K, int N) {
  int idx = blockIdx.x * blockDim.x + threadIdx.x;
  int KT = K >> 5;
  int total = (N >> 4) * KT * 256;
  if (idx >= total) return;
  int v    = idx & 7;
  int lane = (idx >> 3) & 31;
  int t    = idx >> 8;
  int kt = t % KT, nt = t / KT;
  int n    = (nt << 4) + (lane & 15);
  int half = lane >> 4;
  int g = v & 3, blk = v >> 2;
  int k = (kt << 5) + (blk << 4) + (half << 3) + (g << 1);
  union { _Float16 h[2]; unsigned int u; } cv;
  cv.h[0] = (_Float16)B[(size_t)k * N + n];
  cv.h[1] = (_Float16)B[(size_t)(k + 1) * N + n];
  Bp[idx] = cv.u;
}

// Issue one TDM load of a 16-row x 16-dword (32-half) K-slice of A into LDS,
// with 1-dword padding every 16 dwords (17-dword row stride in LDS).
// Descriptor layout per CDNA5 ISA.
__device__ __forceinline__ void tdm_load_tile(unsigned long long gbase, int kt,
                                              unsigned lds_addr, int K) {
  unsigned long long ga = gbase + (unsigned)(kt << 6);       // 64B per K-step
  v4u g0;
  g0[0] = 1u;                                                // count=1, user mode
  g0[1] = lds_addr;                                          // lds_addr (bytes)
  g0[2] = (unsigned)ga;                                      // global_addr[31:0]
  g0[3] = (unsigned)((ga >> 32) & 0x01FFFFFFu) | (2u << 30); // [56:32] | type=2
  v8i g1;
  g1[0] = (int)((2u << 16) | (1u << 20) | (3u << 22));       // data=4B, pad 1dw/16dw
  g1[1] = (int)(((unsigned)(K >> 1) & 0xFFFFu) << 16);       // tensor_dim0 lo16 (dwords)
  g1[2] = 0;                                                 // td0 hi16 | td1 lo16
  g1[3] = (int)((16u << 16) | 0x10u);                        // tile_dim0=16 | td1 hi16 (big)
  g1[4] = 16;                                                // tile_dim1=16, tile_dim2=0
  g1[5] = (int)(K >> 1);                                     // dim0 stride lo32 (dwords)
  g1[6] = 0;                                                 // stride0 hi | stride1 lo
  g1[7] = 0;                                                 // stride1 hi
  v4i zz = {};
#if __clang_major__ >= 23
  v8i zz8 = {};
  __builtin_amdgcn_tensor_load_to_lds(g0, g1, zz, zz, zz8, 0);
#else
  __builtin_amdgcn_tensor_load_to_lds(g0, g1, zz, zz, 0);
#endif
}

// ---------------------------------------------------------------------------
// WMMA GEMM with double-buffered TDM-staged A tiles:
//   C[M,Nmat] = A[M,K](f16,row-major) x Bpacked ; Nmat = Nt*16
// Block = 8 waves sharing one 16-row A tile. Wave 0 DMAs tile kt+1 into the
// alternate LDS buffer *before* the compute of tile kt, so the tensor DMA
// overlaps the ds_load + global B-stream + WMMA; it waits TENSORcnt only
// after the compute. One workgroup barrier per K-step.
// epilogue: (+bias) -> (*gamma*bnmul + beta) -> relu ; writes f32 and/or f16
// ---------------------------------------------------------------------------
__global__ __launch_bounds__(256)
void k_wmma_gemm(const _Float16* __restrict__ A, const unsigned int* __restrict__ Bp,
                 float* __restrict__ C32, int ldC, int colOff,
                 _Float16* __restrict__ C16, int ldC16,
                 int M, int K, int Nt,
                 const float* __restrict__ bias, const float* __restrict__ gamma,
                 const float* __restrict__ beta, float bnmul, int relu) {
  __shared__ unsigned smemA[2][16 * 17];         // 2 x (16 rows x 17-dword stride)
  const int wid  = threadIdx.x >> 5;
  const int lane = threadIdx.x & 31;
  const int ntg_per = Nt >> 3;                   // 8 N-tiles per block
  const int mt  = blockIdx.x / ntg_per;
  const int ntg = blockIdx.x % ntg_per;
  const int nt  = (ntg << 3) + wid;
  const int KT  = K >> 5;
  const int half = lane >> 4;

  v8f acc = {};
  const unsigned int* brow = Bp + ((size_t)nt * KT * 32 + lane) * 8;
  const unsigned lds0 = (unsigned)(size_t)(void*)&smemA[0][0];
  const unsigned lds1 = (unsigned)(size_t)(void*)&smemA[1][0];
  const unsigned long long gbase =
      (unsigned long long)(size_t)A + (size_t)(mt << 4) * K * 2;

  // prologue: fill buffer 0 with tile 0
  if (wid == 0) {
    tdm_load_tile(gbase, 0, lds0, K);
    __builtin_amdgcn_s_wait_tensorcnt(0);
  }
  __syncthreads();

  for (int kt = 0; kt < KT; ++kt) {
    const int cur = kt & 1;
    // kick off next tile's DMA into the other buffer (overlaps compute)
    if (wid == 0 && kt + 1 < KT)
      tdm_load_tile(gbase, kt + 1, cur ? lds0 : lds1, K);

    // A fragment from LDS: row lane%16, 17-dword padded stride
    const unsigned* sp = &smemA[cur][(lane & 15) * 17];
    v8u au;
#pragma unroll
    for (int i = 0; i < 4; ++i) {
      au[i]     = sp[(half << 2) + i];
      au[i + 4] = sp[8 + (half << 2) + i];
    }
    // B fragment: 8 contiguous dwords per lane from the packed buffer
    v8u bu = *(const v8u*)brow;
    __builtin_prefetch(brow + 256, 0, 1);         // next K-step's packed B
    brow += 256;

    v16h av = __builtin_bit_cast(v16h, au);
    v16h bv = __builtin_bit_cast(v16h, bu);
    acc = __builtin_amdgcn_wmma_f32_16x16x32_f16(false, av, false, bv,
                                                 (short)0, acc, false, false);

    // DMA for tile kt+1 must be complete (and everyone done reading buffer
    // cur) before the next iteration reads buffer cur^1 / overwrites cur.
    if (wid == 0) __builtin_amdgcn_s_wait_tensorcnt(0);
    __syncthreads();
  }

  // epilogue: lane L, vgpr r -> row mt*16 + half*8 + r, col nt*16 + (L&15)
  int colIdx = (nt << 4) + (lane & 15);
  float b  = bias  ? bias[colIdx]          : 0.f;
  float g  = gamma ? gamma[colIdx] * bnmul : 0.f;
  float be = gamma ? beta[colIdx]          : 0.f;
  int rbase = (mt << 4) + (half << 3);
#pragma unroll
  for (int r = 0; r < 8; ++r) {
    float v = acc[r] + b;
    if (gamma) v = v * g + be;
    if (relu)  v = fmaxf(v, 0.f);
    if (C32) C32[(size_t)(rbase + r) * ldC + colOff + colIdx] = v;
    if (C16) C16[(size_t)(rbase + r) * ldC16 + colIdx] = (_Float16)v;
  }
}

// ---------------------------------------------------------------------------
// small dense layers (K too small for WMMA)
// ---------------------------------------------------------------------------
__global__ void k_xw1(const float* __restrict__ x, const float* __restrict__ W,
                      float* __restrict__ out, int N) {
  int idx = blockIdx.x * blockDim.x + threadIdx.x;
  if (idx >= N * HF) return;
  int n = idx >> 9, j = idx & (HF - 1);
  float acc = 0.f;
#pragma unroll
  for (int k = 0; k < 9; ++k) acc += x[n * 9 + k] * W[k * HF + j];
  out[idx] = acc;
}

__global__ void k_pca1(const float* __restrict__ x, const float* __restrict__ W,
                       const float* __restrict__ b, const float* __restrict__ g,
                       const float* __restrict__ be, float bnmul,
                       _Float16* __restrict__ out16, int N) {
  int idx = blockIdx.x * blockDim.x + threadIdx.x;
  if (idx >= N * 128) return;
  int n = idx >> 7, j = idx & 127;
  float acc = b[j];
#pragma unroll
  for (int k = 0; k < 9; ++k) acc += x[n * 9 + k] * W[k * 128 + j];
  acc = acc * (g[j] * bnmul) + be[j];
  out16[idx] = (_Float16)fmaxf(acc, 0.f);
}

__global__ void k_scale_enc(const float* __restrict__ scales,
                            const float* __restrict__ sW1, const float* __restrict__ sb1,
                            const float* __restrict__ sW2, const float* __restrict__ sb2,
                            float* __restrict__ hc, int N) {
  int idx = blockIdx.x * blockDim.x + threadIdx.x;
  if (idx >= N * 128) return;
  int n = idx >> 7, j = idx & 127;
  float s = scales[n];
  float acc = sb2[j];
#pragma unroll
  for (int k = 0; k < 32; ++k) {
    float t = fmaxf(s * sW1[k] + sb1[k], 0.f);
    acc += t * sW2[k * 128 + j];
  }
  hc[(size_t)n * COUT + 768 + j] = acc;
}

// ---------------------------------------------------------------------------
// GAT attention pipeline
// ---------------------------------------------------------------------------
__global__ void k_alpha(const float* __restrict__ xw,
                        const float* __restrict__ asrc, const float* __restrict__ adst,
                        float* __restrict__ as, float* __restrict__ ad,
                        float* __restrict__ mmax, float* __restrict__ denom, int N) {
  int t = blockIdx.x * blockDim.x + threadIdx.x;
  if (t >= N * NH) return;
  int n = t >> 2, h = t & 3;
  const float* xp = xw + (size_t)n * HF + h * FH;
  const float* sp = asrc + h * FH;
  const float* dp = adst + h * FH;
  float ss = 0.f, sd = 0.f;
  for (int i = 0; i < FH; ++i) { float v = xp[i]; ss += v * sp[i]; sd += v * dp[i]; }
  as[t] = ss; ad[t] = sd;
  mmax[t] = -__builtin_inff();
  denom[t] = 0.f;
}

__device__ __forceinline__ void atom_max_f32(float* a, float v) {
  if (v >= 0.f) atomicMax((int*)a, __float_as_int(v));
  else          atomicMin((unsigned int*)a, __float_as_uint(v));
}

__device__ __forceinline__ void edge_sd(const int* ei, int e, int E, int* s, int* d) {
  if (e < E) { *s = ei[e]; *d = ei[E + e]; } else { *s = *d = e - E; }
}

__global__ void k_edge_max(const int* __restrict__ ei, int E, int ET,
                           const float* __restrict__ as, const float* __restrict__ ad,
                           float* __restrict__ mmax) {
  int e = blockIdx.x * blockDim.x + threadIdx.x;
  if (e >= ET) return;
  int s, d; edge_sd(ei, e, E, &s, &d);
#pragma unroll
  for (int h = 0; h < NH; ++h) {
    float v = as[s * NH + h] + ad[d * NH + h];
    v = v >= 0.f ? v : 0.2f * v;
    atom_max_f32(&mmax[d * NH + h], v);
  }
}

__global__ void k_edge_expsum(const int* __restrict__ ei, int E, int ET,
                              const float* __restrict__ as, const float* __restrict__ ad,
                              const float* __restrict__ mmax,
                              float* __restrict__ ex, float* __restrict__ denom) {
  int e = blockIdx.x * blockDim.x + threadIdx.x;
  if (e >= ET) return;
  int s, d; edge_sd(ei, e, E, &s, &d);
#pragma unroll
  for (int h = 0; h < NH; ++h) {
    float v = as[s * NH + h] + ad[d * NH + h];
    v = v >= 0.f ? v : 0.2f * v;
    float exv = __expf(v - mmax[d * NH + h]);
    ex[(size_t)e * NH + h] = exv;
    atomicAdd(&denom[d * NH + h], exv);
  }
}

__global__ __launch_bounds__(256)
void k_edge_agg(const int* __restrict__ ei, int E, int ET,
                const float* __restrict__ ex, const float* __restrict__ denom,
                const float* __restrict__ xw, float* __restrict__ out) {
  int e = blockIdx.x;
  if (e >= ET) return;
  int s, d; edge_sd(ei, e, E, &s, &d);
  __shared__ float sal[NH];
  if (threadIdx.x < NH) {
    int h = threadIdx.x;
    sal[h] = ex[(size_t)e * NH + h] / (denom[d * NH + h] + 1e-16f);
  }
  __syncthreads();
  const float* src = xw + (size_t)s * HF;
  float* dst = out + (size_t)d * HF;
  for (int f = threadIdx.x; f < HF; f += blockDim.x)
    atomicAdd(&dst[f], src[f] * sal[f >> 7]);
}

__global__ void k_bias_relu_f16(const float* __restrict__ agg, const float* __restrict__ b,
                                _Float16* __restrict__ out16, int N) {
  int idx = blockIdx.x * blockDim.x + threadIdx.x;
  if (idx >= N * HF) return;
  out16[idx] = (_Float16)fmaxf(agg[idx] + b[idx & (HF - 1)], 0.f);
}

// ---------------------------------------------------------------------------
// GCN
// ---------------------------------------------------------------------------
__global__ void k_deg(const int* __restrict__ ei, int E, int ET, float* __restrict__ deg) {
  int e = blockIdx.x * blockDim.x + threadIdx.x;
  if (e >= ET) return;
  int s, d; edge_sd(ei, e, E, &s, &d);
  (void)s;
  atomicAdd(&deg[d], 1.f);
}

__global__ void k_dis(float* __restrict__ deg, int N) {
  int n = blockIdx.x * blockDim.x + threadIdx.x;
  if (n >= N) return;
  float v = deg[n];
  deg[n] = v > 0.f ? rsqrtf(v) : 0.f;
}

__global__ __launch_bounds__(256)
void k_gcn_agg(const int* __restrict__ ei, int E, int ET,
               const float* __restrict__ dis, const float* __restrict__ xw,
               float* __restrict__ out) {
  int e = blockIdx.x;
  if (e >= ET) return;
  int s, d; edge_sd(ei, e, E, &s, &d);
  float norm = dis[s] * dis[d];
  const float* src = xw + (size_t)s * HF;
  float* dst = out + (size_t)d * HF;
  for (int f = threadIdx.x; f < HF; f += blockDim.x)
    atomicAdd(&dst[f], src[f] * norm);
}

__global__ void k_topo_bias(const float* __restrict__ agg, const float* __restrict__ b3,
                            float* __restrict__ hc, int N) {
  int idx = blockIdx.x * blockDim.x + threadIdx.x;
  if (idx >= N * HF) return;
  int n = idx >> 9, f = idx & (HF - 1);
  hc[(size_t)n * COUT + f] = agg[idx] + b3[f];
}

// ---------------------------------------------------------------------------
// pooling
// ---------------------------------------------------------------------------
__global__ void k_count(const int* __restrict__ batch, float* __restrict__ counts, int N) {
  int n = blockIdx.x * blockDim.x + threadIdx.x;
  if (n >= N) return;
  atomicAdd(&counts[batch[n]], 1.f);
}

__global__ void k_pool_sum(const float* __restrict__ hc, const int* __restrict__ batch,
                           float* __restrict__ frag, int N) {
  int idx = blockIdx.x * blockDim.x + threadIdx.x;
  if (idx >= N * COUT) return;
  int n = idx / COUT, f = idx - n * COUT;
  atomicAdd(&frag[(size_t)batch[n] * COUT + f], hc[idx]);
}

__global__ void k_pool_div(float* __restrict__ frag, const float* __restrict__ counts, int G) {
  int idx = blockIdx.x * blockDim.x + threadIdx.x;
  if (idx >= G * COUT) return;
  int g = idx / COUT;
  frag[idx] /= fmaxf(counts[g], 1.f);
}

// ---------------------------------------------------------------------------
// host orchestration
// ---------------------------------------------------------------------------
extern "C" void kernel_launch(void* const* d_in, const int* in_sizes, int n_in,
                              void* d_out, int out_size, void* d_ws, size_t ws_size,
                              hipStream_t stream) {
  const float* x      = (const float*)d_in[0];
  const int*   ei     = (const int*)d_in[1];
  const int*   batch  = (const int*)d_in[2];
  const float* scales = (const float*)d_in[3];
  const float* W1 = (const float*)d_in[4];
  const float* as1 = (const float*)d_in[5];
  const float* ad1 = (const float*)d_in[6];
  const float* b1 = (const float*)d_in[7];
  const float* W2 = (const float*)d_in[8];
  const float* as2 = (const float*)d_in[9];
  const float* ad2 = (const float*)d_in[10];
  const float* b2 = (const float*)d_in[11];
  const float* W3 = (const float*)d_in[12];
  const float* b3 = (const float*)d_in[13];
  const float* pW1 = (const float*)d_in[14];
  const float* pb1 = (const float*)d_in[15];
  const float* g1  = (const float*)d_in[16];
  const float* be1 = (const float*)d_in[17];
  const float* pW2 = (const float*)d_in[18];
  const float* pb2 = (const float*)d_in[19];
  const float* g2  = (const float*)d_in[20];
  const float* be2 = (const float*)d_in[21];
  const float* pW3 = (const float*)d_in[22];
  const float* pb3 = (const float*)d_in[23];
  const float* sW1 = (const float*)d_in[24];
  const float* sb1 = (const float*)d_in[25];
  const float* sW2 = (const float*)d_in[26];
  const float* sb2 = (const float*)d_in[27];

  const int N  = in_sizes[0] / 9;
  const int E  = in_sizes[1] / 2;
  const int ET = E + N;
  const int G  = out_size / COUT - N;
  const float BNMUL = 1.0f / sqrtf(1.0f + 1e-5f);

  float* outp = (float*)d_out;
  float* frag = outp;                       // [G, 896]
  float* hc   = outp + (size_t)G * COUT;    // [N, 896]

  // workspace carve-out (256B aligned)
  size_t off = 0;
  auto wsalloc = [&](size_t bytes) -> char* {
    off = (off + 255) & ~(size_t)255;
    char* p = (char*)d_ws + off;
    off += bytes;
    return p;
  };
  float*        bufA   = (float*)wsalloc((size_t)N * HF * 4);    // xw (ping)
  float*        bufB   = (float*)wsalloc((size_t)N * HF * 4);    // aggregation (pong)
  _Float16*     hf16   = (_Float16*)wsalloc((size_t)N * HF * 2); // f16 activations
  float*        aS     = (float*)wsalloc((size_t)N * NH * 4);
  float*        aD     = (float*)wsalloc((size_t)N * NH * 4);
  float*        mmax   = (float*)wsalloc((size_t)N * NH * 4);
  float*        denom  = (float*)wsalloc((size_t)N * NH * 4);
  float*        ex     = (float*)wsalloc((size_t)ET * NH * 4);
  float*        deg    = (float*)wsalloc((size_t)N * 4);
  _Float16*     pca1_16 = (_Float16*)wsalloc((size_t)N * 128 * 2);
  _Float16*     pca2_16 = (_Float16*)wsalloc((size_t)N * 256 * 2);
  float*        counts = (float*)wsalloc((size_t)G * 4);
  unsigned int* W2p  = (unsigned int*)wsalloc((size_t)512 * 512 * 2);
  unsigned int* W3p  = (unsigned int*)wsalloc((size_t)512 * 512 * 2);
  unsigned int* pW2p = (unsigned int*)wsalloc((size_t)128 * 256 * 2);
  unsigned int* pW3p = (unsigned int*)wsalloc((size_t)256 * 256 * 2);
  (void)ws_size;

  const int T = 256;
  auto zero = [&](float* p, long long n) {
    k_fill_zero<<<cdiv_ll(n, T), T, 0, stream>>>(p, n);
  };

  // 0. pack weights into WMMA B layout (f16)
  k_pack_b<<<cdiv_ll((size_t)512 * 512 / 2, T), T, 0, stream>>>(W2, W2p, 512, 512);
  k_pack_b<<<cdiv_ll((size_t)512 * 512 / 2, T), T, 0, stream>>>(W3, W3p, 512, 512);
  k_pack_b<<<cdiv_ll((size_t)128 * 256 / 2, T), T, 0, stream>>>(pW2, pW2p, 128, 256);
  k_pack_b<<<cdiv_ll((size_t)256 * 256 / 2, T), T, 0, stream>>>(pW3, pW3p, 256, 256);

  // ---- GAT layer 1 ----
  k_xw1<<<cdiv_ll((long long)N * HF, T), T, 0, stream>>>(x, W1, bufA, N);
  k_alpha<<<cdiv_ll((long long)N * NH, T), T, 0, stream>>>(bufA, as1, ad1, aS, aD, mmax, denom, N);
  zero(bufB, (long long)N * HF);
  k_edge_max<<<cdiv_ll(ET, T), T, 0, stream>>>(ei, E, ET, aS, aD, mmax);
  k_edge_expsum<<<cdiv_ll(ET, T), T, 0, stream>>>(ei, E, ET, aS, aD, mmax, ex, denom);
  k_edge_agg<<<ET, T, 0, stream>>>(ei, E, ET, ex, denom, bufA, bufB);
  k_bias_relu_f16<<<cdiv_ll((long long)N * HF, T), T, 0, stream>>>(bufB, b1, hf16, N);

  // ---- GAT layer 2: xw2 = h1 @ W2 (WMMA, double-buffered TDM A) ----
  k_wmma_gemm<<<(N / 16) * (32 / 8), T, 0, stream>>>(
      hf16, W2p, bufA, HF, 0, nullptr, 0, N, 512, 32,
      nullptr, nullptr, nullptr, 1.f, 0);
  k_alpha<<<cdiv_ll((long long)N * NH, T), T, 0, stream>>>(bufA, as2, ad2, aS, aD, mmax, denom, N);
  zero(bufB, (long long)N * HF);
  k_edge_max<<<cdiv_ll(ET, T), T, 0, stream>>>(ei, E, ET, aS, aD, mmax);
  k_edge_expsum<<<cdiv_ll(ET, T), T, 0, stream>>>(ei, E, ET, aS, aD, mmax, ex, denom);
  k_edge_agg<<<ET, T, 0, stream>>>(ei, E, ET, ex, denom, bufA, bufB);
  k_bias_relu_f16<<<cdiv_ll((long long)N * HF, T), T, 0, stream>>>(bufB, b2, hf16, N);

  // ---- GCN: xw3 = h2 @ W3 (WMMA, double-buffered TDM A), normalized aggregate ----
  k_wmma_gemm<<<(N / 16) * (32 / 8), T, 0, stream>>>(
      hf16, W3p, bufA, HF, 0, nullptr, 0, N, 512, 32,
      nullptr, nullptr, nullptr, 1.f, 0);
  zero(deg, N);
  k_deg<<<cdiv_ll(ET, T), T, 0, stream>>>(ei, E, ET, deg);
  k_dis<<<cdiv_ll(N, T), T, 0, stream>>>(deg, N);
  zero(bufB, (long long)N * HF);
  k_gcn_agg<<<ET, T, 0, stream>>>(ei, E, ET, deg, bufA, bufB);
  k_topo_bias<<<cdiv_ll((long long)N * HF, T), T, 0, stream>>>(bufB, b3, hc, N);

  // ---- PCA encoder ----
  k_pca1<<<cdiv_ll((long long)N * 128, T), T, 0, stream>>>(x, pW1, pb1, g1, be1, BNMUL, pca1_16, N);
  k_wmma_gemm<<<(N / 16) * (16 / 8), T, 0, stream>>>(
      pca1_16, pW2p, nullptr, 0, 0, pca2_16, 256, N, 128, 16,
      pb2, g2, be2, BNMUL, 1);
  k_wmma_gemm<<<(N / 16) * (16 / 8), T, 0, stream>>>(
      pca2_16, pW3p, hc, COUT, 512, nullptr, 0, N, 256, 16,
      pb3, nullptr, nullptr, 1.f, 0);

  // ---- scale encoder ----
  k_scale_enc<<<cdiv_ll((long long)N * 128, T), T, 0, stream>>>(scales, sW1, sb1, sW2, sb2, hc, N);

  // ---- segment-mean pooling ----
  zero(frag, (long long)G * COUT);
  zero(counts, G);
  k_count<<<cdiv_ll(N, T), T, 0, stream>>>(batch, counts, N);
  k_pool_sum<<<cdiv_ll((long long)N * COUT, T), T, 0, stream>>>(hc, batch, frag, N);
  k_pool_div<<<cdiv_ll((long long)G * COUT, T), T, 0, stream>>>(frag, counts, G);
}
